// BasicMultiHeadSelfAttention_60181081751873
// MI455X (gfx1250) — matrix operand order repaired and checked by reading, not compiled, not run on hardware
//
#include <hip/hip_runtime.h>

#define B_ 4
#define S_ 2048
#define E_ 1024
#define H_ 16
#define HD_ 64

#if defined(__has_builtin)
#  if __has_builtin(__builtin_amdgcn_global_load_async_to_lds_b128) && \
      __has_builtin(__builtin_amdgcn_s_wait_asynccnt)
#    define USE_ASYNC 1
#  endif
#endif
#ifndef USE_ASYNC
#  define USE_ASYNC 0
#endif

typedef __attribute__((ext_vector_type(16))) __bf16 v16bf;
typedef __attribute__((ext_vector_type(8)))  __bf16 v8bf;
typedef __attribute__((ext_vector_type(8)))  float  v8f;
typedef __attribute__((ext_vector_type(4)))  int    v4i;

__device__ __forceinline__ unsigned short f2bf(float f) {
  unsigned int u = __float_as_uint(f);
  u += 0x7FFFu + ((u >> 16) & 1u);          // round-to-nearest-even
  return (unsigned short)(u >> 16);
}

#if USE_ASYNC
__device__ __forceinline__ void async_cp_b128(const unsigned short* g, unsigned short* l) {
  __builtin_amdgcn_global_load_async_to_lds_b128((v4i*)g, (v4i*)l, 0, 0);
}
#endif

// A-matrix fragment (16x32 bf16): per lane, 16 consecutive K elements
__device__ __forceinline__ v16bf load_frag_a(const unsigned short* p) {
  v8bf lo = *(const v8bf*)(p);
  v8bf hi = *(const v8bf*)(p + 8);
  return __builtin_shufflevector(lo, hi, 0,1,2,3,4,5,6,7,8,9,10,11,12,13,14,15);
}
// B-matrix fragment (32x16 bf16): per lane, 16 consecutive K elements
__device__ __forceinline__ v16bf load_frag_b(const unsigned short* p) {
  v8bf lo = *(const v8bf*)(p);
  v8bf hi = *(const v8bf*)(p + 8);
  return __builtin_shufflevector(lo, hi, 0,1,2,3,4,5,6,7,8,9,10,11,12,13,14,15);
}

__device__ __forceinline__ v8f wmma_bf16(v16bf a, v16bf b, v8f c) {
  return __builtin_amdgcn_wmma_f32_16x16x32_bf16(false, a, false, b, (short)0, c,
                                                 false, false);
}

// ---------------- conversion kernels ----------------
__global__ void k_cvt(const float* __restrict__ x, unsigned short* __restrict__ y, int n8) {
  int i = blockIdx.x * blockDim.x + threadIdx.x;
  if (i < n8) {
    const float* xp = x + (size_t)i * 8;
    union { v8bf v; unsigned short u[8]; } o;
    #pragma unroll
    for (int j = 0; j < 8; j++) o.u[j] = f2bf(xp[j]);
    *(v8bf*)(y + (size_t)i * 8) = o.v;
  }
}

// wt[n*rows + k] = w[k*cols + n] via 32x32 LDS tile (coalesced both sides)
__global__ __launch_bounds__(256)
void k_transpose_cvt(const float* __restrict__ w, unsigned short* __restrict__ wt,
                     int rows, int cols) {
  __shared__ unsigned short tile[32][33];
  const int c0 = blockIdx.x * 32, r0 = blockIdx.y * 32;
  const int tx = threadIdx.x & 31, ty = threadIdx.x >> 5;   // 32 x 8
  #pragma unroll
  for (int i = 0; i < 32; i += 8)
    tile[ty + i][tx] = f2bf(w[(size_t)(r0 + ty + i) * cols + c0 + tx]);
  __syncthreads();
  #pragma unroll
  for (int i = 0; i < 32; i += 8)
    wt[(size_t)(c0 + ty + i) * rows + r0 + tx] = tile[tx][ty + i];
}

// ---------------- bf16 GEMM: C = A[M,K] * Bt[N,K]^T + bias ----------------
// Block tile 128x256, 8 waves each 64x64, BK=32, double-buffered async staging.
// mode 0: f32 row-major output.  mode 1: scatter bf16 into q/k/v head layout.
#define BM 128
#define BN 256
#define LDT 48

__global__ __launch_bounds__(256)
void k_gemm_bf16(const unsigned short* __restrict__ A,
                 const unsigned short* __restrict__ Bt,
                 const float* __restrict__ bias,
                 float* __restrict__ Cout,
                 unsigned short* __restrict__ q_out,
                 unsigned short* __restrict__ k_out,
                 unsigned short* __restrict__ v_out,
                 int M, int N, int K, int mode)
{
  __shared__ __align__(16) unsigned short Al[2][BM * LDT];
  __shared__ __align__(16) unsigned short Bl[2][BN * LDT];

  const int tid  = threadIdx.x;
  const int lane = tid & 31;
  const int wid  = tid >> 5;
  const int wm   = wid >> 2;        // 0..1 (64 rows)
  const int wn   = wid & 3;         // 0..3 (64 cols)
  const int hi   = lane >> 4;
  const int ln   = lane & 15;
  const int blockM = blockIdx.x * BM;
  const int blockN = blockIdx.y * BN;

  v8f acc[4][4];
  #pragma unroll
  for (int i = 0; i < 4; i++)
    #pragma unroll
    for (int j = 0; j < 4; j++) acc[i][j] = {};

  auto stage = [&](int buf, int k0) {
#if USE_ASYNC
    #pragma unroll
    for (int i = 0; i < 2; i++) {                    // A: 512 b128 chunks
      int c = (wid * 2 + i) * 32 + lane;
      int row = c >> 2, seg = c & 3;
      async_cp_b128(A + (size_t)(blockM + row) * K + k0 + seg * 8,
                    &Al[buf][row * LDT + seg * 8]);
    }
    #pragma unroll
    for (int i = 0; i < 4; i++) {                    // B: 1024 b128 chunks
      int c = (wid * 4 + i) * 32 + lane;
      int row = c >> 2, seg = c & 3;
      async_cp_b128(Bt + (size_t)(blockN + row) * K + k0 + seg * 8,
                    &Bl[buf][row * LDT + seg * 8]);
    }
#else
    #pragma unroll
    for (int t = 0; t < 2; t++) {
      int c = tid + t * 256;
      int row = c >> 2, seg = c & 3;
      *(v8bf*)(&Al[buf][row * LDT + seg * 8]) =
          *(const v8bf*)(A + (size_t)(blockM + row) * K + k0 + seg * 8);
    }
    #pragma unroll
    for (int t = 0; t < 4; t++) {
      int c = tid + t * 256;
      int row = c >> 2, seg = c & 3;
      *(v8bf*)(&Bl[buf][row * LDT + seg * 8]) =
          *(const v8bf*)(Bt + (size_t)(blockN + row) * K + k0 + seg * 8);
    }
#endif
  };

  auto compute = [&](int buf) {
    v16bf af[4], bf[4];
    #pragma unroll
    for (int mt = 0; mt < 4; mt++)
      af[mt] = load_frag_a(&Al[buf][(wm * 64 + mt * 16 + ln) * LDT + hi * 16]);
    #pragma unroll
    for (int nt = 0; nt < 4; nt++)
      bf[nt] = load_frag_b(&Bl[buf][(wn * 64 + nt * 16 + ln) * LDT + hi * 16]);
    #pragma unroll
    for (int mt = 0; mt < 4; mt++)
      #pragma unroll
      for (int nt = 0; nt < 4; nt++)
        acc[mt][nt] = wmma_bf16(af[mt], bf[nt], acc[mt][nt]);
  };

  const int KT = K / 32;
  stage(0, 0);
  // steady state: stage(next) overlaps compute(cur); last iteration peeled so
  // the loop body has unconditional waits (no per-iteration branch).
  for (int kt = 0; kt < KT - 1; ++kt) {
    const int cur = kt & 1;
    stage(cur ^ 1, (kt + 1) * 32);
#if USE_ASYNC
    __builtin_amdgcn_s_wait_asynccnt(6);   // retire the older batch only
#endif
    __syncthreads();
    compute(cur);
    __syncthreads();
  }
#if USE_ASYNC
  __builtin_amdgcn_s_wait_asynccnt(0);
#endif
  __syncthreads();
  compute((KT - 1) & 1);

  #pragma unroll
  for (int mt = 0; mt < 4; mt++) {
    #pragma unroll
    for (int nt = 0; nt < 4; nt++) {
      int col = blockN + wn * 64 + nt * 16 + ln;
      float bv = bias[col];
      #pragma unroll
      for (int r = 0; r < 8; r++) {
        int row = blockM + wm * 64 + mt * 16 + r + 8 * hi;
        float v = acc[mt][nt][r] + bv;
        if (mode == 0) {
          Cout[(size_t)row * N + col] = v;
        } else {
          int which = col >> 10;
          int e = col & 1023;
          int h = e >> 6, d = e & 63;
          if (which == 0) v *= 0.125f;     // 1/sqrt(HD)
          unsigned short* dst = (which == 0) ? q_out : (which == 1 ? k_out : v_out);
          size_t off = ((size_t)((row >> 11) * H_ + h) * S_ + (row & (S_ - 1))) * HD_ + d;
          dst[off] = f2bf(v);
        }
      }
    }
  }
}

// ---------------- causal flash attention ----------------
// grid: (S/128, B*H); 8 waves, each owns 16 query rows.
__global__ __launch_bounds__(256)
void k_flash_attn(const unsigned short* __restrict__ Qh,   // [BH][S][64] bf16
                  const unsigned short* __restrict__ Kh,
                  const unsigned short* __restrict__ Vh,
                  unsigned short* __restrict__ joined)     // [B][S][E] bf16
{
  __shared__ __align__(16) unsigned short Kl[32 * 72];
  __shared__ __align__(16) unsigned short Vl[64 * 56];     // transposed [d][key]
  __shared__ __align__(16) unsigned short Pl[8][16 * 48];  // per-wave P tile

  const int tid  = threadIdx.x;
  const int lane = tid & 31;
  const int wid  = tid >> 5;
  const int hi   = lane >> 4;
  const int ln   = lane & 15;
  const int rBase = blockIdx.x * 128;
  const int bh    = blockIdx.y;
  const int b = bh >> 4, h = bh & 15;

  const unsigned short* qp = Qh + ((size_t)bh * S_ + rBase + wid * 16 + ln) * HD_;
  const v16bf qf0 = load_frag_a(qp + hi * 16);
  const v16bf qf1 = load_frag_a(qp + 32 + hi * 16);

  v8f oacc[4];
  #pragma unroll
  for (int t = 0; t < 4; t++) oacc[t] = {};
  float mrow[8], lrow[8];
  #pragma unroll
  for (int r = 0; r < 8; r++) { mrow[r] = -1e30f; lrow[r] = 0.0f; }

  const int nblk = (rBase + 128) / 32;     // causal: stop at the diagonal
  for (int jb = 0; jb < nblk; jb++) {
    const int jBase = jb * 32;
    __syncthreads();
    {
#if USE_ASYNC
      int c = wid * 32 + lane;             // 256 b128 chunks for K block
      int key = c >> 3, seg = c & 7;
      async_cp_b128(Kh + ((size_t)bh * S_ + jBase + key) * HD_ + seg * 8,
                    &Kl[key * 72 + seg * 8]);
#else
      int key = tid >> 3, seg = tid & 7;
      *(v8bf*)(&Kl[key * 72 + seg * 8]) =
          *(const v8bf*)(Kh + ((size_t)bh * S_ + jBase + key) * HD_ + seg * 8);
#endif
      int keyv = tid >> 3, segv = tid & 7;
      union { v8bf v; unsigned short u[8]; } tv;
      tv.v = *(const v8bf*)(Vh + ((size_t)bh * S_ + jBase + keyv) * HD_ + segv * 8);
      #pragma unroll
      for (int i = 0; i < 8; i++) Vl[(segv * 8 + i) * 56 + keyv] = tv.u[i];
    }
#if USE_ASYNC
    __builtin_amdgcn_s_wait_asynccnt(0);
#endif
    __syncthreads();

    v8f sc0 = {}, sc1 = {};
    {
      v16bf kb;
      kb = load_frag_b(&Kl[(ln) * 72 + 0  + hi * 16]); sc0 = wmma_bf16(qf0, kb, sc0);
      kb = load_frag_b(&Kl[(ln) * 72 + 32 + hi * 16]); sc0 = wmma_bf16(qf1, kb, sc0);
      kb = load_frag_b(&Kl[(16 + ln) * 72 + 0  + hi * 16]); sc1 = wmma_bf16(qf0, kb, sc1);
      kb = load_frag_b(&Kl[(16 + ln) * 72 + 32 + hi * 16]); sc1 = wmma_bf16(qf1, kb, sc1);
    }

    const int myq = rBase + wid * 16;
    float p0[8], p1[8], scale[8];
    #pragma unroll
    for (int r = 0; r < 8; r++) {
      const int row = myq + r + 8 * hi;
      float s0 = sc0[r], s1 = sc1[r];
      if (jBase + ln      > row) s0 = -1e30f;
      if (jBase + 16 + ln > row) s1 = -1e30f;
      float mx = fmaxf(s0, s1);
      #pragma unroll
      for (int m = 1; m < 16; m <<= 1) mx = fmaxf(mx, __shfl_xor(mx, m, 32));
      float mn = fmaxf(mrow[r], mx);
      scale[r] = __expf(mrow[r] - mn);
      s0 = __expf(s0 - mn);
      s1 = __expf(s1 - mn);
      float rs = s0 + s1;
      #pragma unroll
      for (int m = 1; m < 16; m <<= 1) rs += __shfl_xor(rs, m, 32);
      lrow[r] = lrow[r] * scale[r] + rs;
      mrow[r] = mn;
      p0[r] = s0; p1[r] = s1;
    }
    #pragma unroll
    for (int t = 0; t < 4; t++)
      #pragma unroll
      for (int r = 0; r < 8; r++) oacc[t][r] *= scale[r];

    unsigned short* Pw = Pl[wid];
    #pragma unroll
    for (int r = 0; r < 8; r++) {
      int row = r + 8 * hi;
      Pw[row * 48 + ln]      = f2bf(p0[r]);
      Pw[row * 48 + 16 + ln] = f2bf(p1[r]);
    }
    v16bf pa = load_frag_a(&Pw[ln * 48 + hi * 16]);
    #pragma unroll
    for (int t = 0; t < 4; t++) {
      v16bf vb = load_frag_b(&Vl[(t * 16 + ln) * 56 + hi * 16]);
      oacc[t] = wmma_bf16(pa, vb, oacc[t]);
    }
  }

  #pragma unroll
  for (int r = 0; r < 8; r++) {
    float inv = 1.0f / lrow[r];
    int s = rBase + wid * 16 + r + 8 * hi;
    #pragma unroll
    for (int t = 0; t < 4; t++) {
      int col = h * HD_ + t * 16 + ln;
      joined[((size_t)b * S_ + s) * E_ + col] = f2bf(oacc[t][r] * inv);
    }
  }
}

// ---------------- launcher ----------------
extern "C" void kernel_launch(void* const* d_in, const int* in_sizes, int n_in,
                              void* d_out, int out_size, void* d_ws, size_t ws_size,
                              hipStream_t stream) {
  (void)in_sizes; (void)n_in; (void)out_size; (void)ws_size;
  const float* Q  = (const float*)d_in[0];
  const float* Wp = (const float*)d_in[1];
  const float* bp = (const float*)d_in[2];
  const float* Wo = (const float*)d_in[3];
  const float* bo = (const float*)d_in[4];
  float* out = (float*)d_out;

  const size_t M = (size_t)B_ * S_;                 // 8192
  unsigned short* ws     = (unsigned short*)d_ws;
  unsigned short* Qbf    = ws;                                   // M*E
  unsigned short* WpT    = Qbf  + M * E_;                        // 3E*E
  unsigned short* WoT    = WpT  + (size_t)3 * E_ * E_;           // E*E
  unsigned short* qh     = WoT  + (size_t)E_ * E_;               // BH*S*HD
  unsigned short* kh     = qh   + (size_t)B_ * H_ * S_ * HD_;
  unsigned short* vh     = kh   + (size_t)B_ * H_ * S_ * HD_;
  unsigned short* joined = vh   + (size_t)B_ * H_ * S_ * HD_;    // M*E

  int nQ8 = (int)(M * E_ / 8);
  k_cvt<<<(nQ8 + 255) / 256, 256, 0, stream>>>(Q, Qbf, nQ8);
  dim3 gt1((3 * E_) / 32, E_ / 32);
  k_transpose_cvt<<<gt1, 256, 0, stream>>>(Wp, WpT, E_, 3 * E_);
  dim3 gt2(E_ / 32, E_ / 32);
  k_transpose_cvt<<<gt2, 256, 0, stream>>>(Wo, WoT, E_, E_);

  dim3 g1((unsigned)(M / BM), (unsigned)((3 * E_) / BN));
  k_gemm_bf16<<<g1, 256, 0, stream>>>(Qbf, WpT, bp, nullptr, qh, kh, vh,
                                      (int)M, 3 * E_, E_, 1);

  dim3 g2(S_ / 128, B_ * H_);
  k_flash_attn<<<g2, 256, 0, stream>>>(qh, kh, vh, joined);

  dim3 g3((unsigned)(M / BM), E_ / BN);
  k_gemm_bf16<<<g3, 256, 0, stream>>>(joined, WoT, bo, out,
                                      nullptr, nullptr, nullptr,
                                      (int)M, E_, E_, 0);
}
